// MoEGate_1297080124195
// MI455X (gfx1250) — matrix operand order repaired and checked by reading, not compiled, not run on hardware
//
#include <hip/hip_runtime.h>

// ---- CDNA5 WMMA / native vector types ---------------------------------------
typedef __attribute__((ext_vector_type(16))) __bf16 v16bf;
typedef __attribute__((ext_vector_type(8)))  float  v8f;
typedef __attribute__((ext_vector_type(4)))  float  f32x4;   // native vec: OK for
                                                             // __builtin_nontemporal_load

#define HDIM   2048
#define NEXP   64
#define KSTEP  32
#define NITER  (HDIM / KSTEP)                 // 64
#define WAVES  4
#define BLOCK  (WAVES * 32)
#define TOK_PER_WAVE 16
#define TOK_PER_WG   (WAVES * TOK_PER_WAVE)   // 64

// LDS: double-buffered bf16 weight chunk, 80 bytes/expert row (64 data + 16 pad
// -> 20-dword stride, conflict-free across 16 lanes, 16 B aligned), plus
// per-wave logits scratch padded to 66 floats/token (66 % 64 = 2 -> no
// conflicts on the strided epilogue reads).
#define BROW_BYTES 80
#define LROW       66

union BfPack16 { __bf16 h[16]; uint4 u[2]; v16bf v; };

__device__ __forceinline__ void cvt8_f32_bf16(const f32x4& f0, const f32x4& f1,
                                              __bf16* dst) {
  dst[0] = (__bf16)f0[0]; dst[1] = (__bf16)f0[1];
  dst[2] = (__bf16)f0[2]; dst[3] = (__bf16)f0[3];
  dst[4] = (__bf16)f1[0]; dst[5] = (__bf16)f1[1];
  dst[6] = (__bf16)f1[2]; dst[7] = (__bf16)f1[3];
}

__global__ __launch_bounds__(BLOCK)
void moe_gate_wmma_kernel(const float* __restrict__ x,      // [T, 2048]
                          const float* __restrict__ wgt,    // [64, 2048]
                          int*   __restrict__ out_idx,      // [T, 2]
                          float* __restrict__ out_w,        // [T, 2]
                          int T) {
  __shared__ __align__(16) unsigned char ldsB[2][NEXP * BROW_BYTES]; // 10240 B
  __shared__ float ldsLog[WAVES * TOK_PER_WAVE * LROW];              // 16896 B

  const int tid    = threadIdx.x;
  const int lane   = tid & 31;
  const int wave   = tid >> 5;
  const int lane16 = lane & 15;
  const bool lo    = (lane < 16);
  const int wgTok  = blockIdx.x * TOK_PER_WG;

  v8f acc[4] = {};   // 16 tokens x 64 experts, fp32 accumulators

  // A-fragment addressing (16-bit A layout, 16x32):
  //   lanes 0-15 : VGPR0-3 = K0..7,  VGPR4-7 = K16..23   (row M = lane)
  //   lanes16-31 : VGPR0-3 = K8..15, VGPR4-7 = K24..31   (row M = lane-16)
  const int arow = wgTok + wave * TOK_PER_WAVE + lane16;
  const float* aBase = x + (size_t)arow * HDIM + (lo ? 0 : 8);
  const int bByte = lo ? 0 : 16;  // B 32x16: lanes 0-15 K0-7,K16-23; lanes 16-31 K8-15,K24-31

  // Weight staging assignment: 128 threads x 16 floats = one 64x32 chunk.
  const int eSt    = tid >> 1;            // 0..63
  const int koffSt = (tid & 1) * 16;      // 0 or 16
  const float* wSt = wgt + (size_t)eSt * HDIM + koffSt;
  uint4* stDst[2] = {
      (uint4*)(ldsB[0] + eSt * BROW_BYTES + koffSt * 2),
      (uint4*)(ldsB[1] + eSt * BROW_BYTES + koffSt * 2)};

  // ---- prologue: stage chunk 0 into buffer 0 --------------------------------
  {
    const f32x4* wp = (const f32x4*)wSt;
    f32x4 f0 = wp[0], f1 = wp[1], f2 = wp[2], f3 = wp[3];
    BfPack16 p;
    cvt8_f32_bf16(f0, f1, p.h);
    cvt8_f32_bf16(f2, f3, p.h + 8);
    stDst[0][0] = p.u[0];
    stDst[0][1] = p.u[1];
  }

#pragma unroll 2
  for (int it = 0; it < NITER; ++it) {
    const int kk = it * KSTEP;
    __syncthreads();   // staged buffer (it & 1) now visible; other buffer free

    // ---- issue NEXT chunk's global loads early (overlap with compute) ------
    // Branchless: last iteration re-reads a valid chunk and stores to the
    // dead buffer; harmless.
    const int kkNext = (kk + KSTEP < HDIM) ? (kk + KSTEP) : kk;
    const f32x4* wp = (const f32x4*)(wSt + kkNext);
    f32x4 w0 = wp[0], w1 = wp[1], w2 = wp[2], w3 = wp[3];

    // ---- A fragment: 2 x 32B streaming (non-temporal) global loads ---------
    const float* ap = aBase + kk;
    f32x4 a0 = __builtin_nontemporal_load((const f32x4*)ap);
    f32x4 a1 = __builtin_nontemporal_load((const f32x4*)ap + 1);
    f32x4 a2 = __builtin_nontemporal_load((const f32x4*)(ap + 16));
    f32x4 a3 = __builtin_nontemporal_load((const f32x4*)(ap + 16) + 1);

    // ---- preload ALL 4 B fragments into distinct regs (one DS wait) --------
    BfPack16 B[4];
#pragma unroll
    for (int n = 0; n < 4; ++n) {
      const int e = n * 16 + lane16;
      const uint4* bp = (const uint4*)(ldsB[it & 1] + e * BROW_BYTES + bByte);
      B[n].u[0] = bp[0];
      B[n].u[1] = bp[2];
    }

    BfPack16 A;
    cvt8_f32_bf16(a0, a1, A.h);
    cvt8_f32_bf16(a2, a3, A.h + 8);

    // ---- 4 back-to-back WMMAs (independent accumulators, shared A) ---------
#pragma unroll
    for (int n = 0; n < 4; ++n) {
      acc[n] = __builtin_amdgcn_wmma_f32_16x16x32_bf16(
          false, A.v, false, B[n].v, (short)0, acc[n], false, false);
    }

    // ---- convert + store NEXT chunk into the other buffer ------------------
    {
      BfPack16 p;
      cvt8_f32_bf16(w0, w1, p.h);
      cvt8_f32_bf16(w2, w3, p.h + 8);
      uint4* dst = stDst[(it + 1) & 1];
      dst[0] = p.u[0];
      dst[1] = p.u[1];
    }
  }

  // ---- epilogue: scatter logits to LDS, one lane per token ------------------
  // C/D layout: lanes 0-15 -> N=lane, M=r ; lanes 16-31 -> N=lane-16, M=8+r
  float* lbase = ldsLog + wave * (TOK_PER_WAVE * LROW);
#pragma unroll
  for (int n = 0; n < 4; ++n) {
#pragma unroll
    for (int r = 0; r < 8; ++r) {
      const int tl = lo ? r : (r + 8);
      lbase[tl * LROW + n * 16 + lane16] = acc[n][r];
    }
  }

  if (lo) {
    const float* row = lbase + lane * LROW;
    // top-2 scan, strict '>' -> lower index wins ties (matches jax top_k)
    float v1 = -__builtin_inff(); int i1 = 0;
    float v2 = -__builtin_inff(); int i2 = 0;
#pragma unroll 8
    for (int e = 0; e < NEXP; ++e) {
      const float v = row[e];
      if (v > v1)      { v2 = v1; i2 = i1; v1 = v; i1 = e; }
      else if (v > v2) { v2 = v;  i2 = e; }
    }
    // softmax sum with max (= v1) subtracted
    float s = 0.0f;
#pragma unroll 8
    for (int e = 0; e < NEXP; ++e) s += __expf(row[e] - v1);
    const float p1 = 1.0f / s;                 // exp(v1 - v1)/s
    const float p2 = __expf(v2 - v1) / s;
    const float inv = 1.0f / (p1 + p2 + 1e-20f);

    const int gt = wgTok + wave * TOK_PER_WAVE + lane;
    out_idx[2 * gt + 0] = i1;
    out_idx[2 * gt + 1] = i2;
    out_w[2 * gt + 0] = p1 * inv;
    out_w[2 * gt + 1] = p2 * inv;
  }
}

extern "C" void kernel_launch(void* const* d_in, const int* in_sizes, int n_in,
                              void* d_out, int out_size, void* d_ws, size_t ws_size,
                              hipStream_t stream) {
  const float* x = (const float*)d_in[0];   // [4,4096,2048] fp32
  const float* w = (const float*)d_in[1];   // [64,2048] fp32
  const int T = in_sizes[0] / HDIM;         // 16384 tokens

  // output: topk_idx (int32, T*2) followed by topk_weight (fp32, T*2)
  int*   out_idx = (int*)d_out;
  float* out_w   = (float*)d_out + (size_t)T * 2;

  const int grid = T / TOK_PER_WG;          // 256 workgroups
  moe_gate_wmma_kernel<<<grid, BLOCK, 0, stream>>>(x, w, out_idx, out_w, T);
}
